// RandLANet_68856915689537
// MI455X (gfx1250) — matrix operand order repaired and checked by reading, not compiled, not run on hardware
//
#include <hip/hip_runtime.h>
#include <hip/hip_bf16.h>

// ---------------------------------------------------------------------------
// CDNA5 (gfx1250) RandLA-Net forward. wave32; WMMA 16x16x32 f16 (f32 acc) for
// all shared-MLP and attention-logit GEMMs, WMMA 16x16x4 f32 for KNN Gram
// tiles (exact f32, matches reference); TDM (tensor_load_to_lds) staging on
// decoder GEMMs.
// ---------------------------------------------------------------------------

typedef __attribute__((ext_vector_type(16))) _Float16 v16h;
typedef __attribute__((ext_vector_type(8)))  float    v8f;
typedef __attribute__((ext_vector_type(4)))  float    v4f;

#ifndef __has_builtin
#define __has_builtin(x) 0
#endif
#if __has_builtin(__builtin_amdgcn_tensor_load_to_lds) && __has_builtin(__builtin_amdgcn_s_wait_tensorcnt)
#define HAVE_TDM 1
#else
#define HAVE_TDM 0
#endif
#if __has_builtin(__builtin_amdgcn_wmma_f32_16x16x4_f32)
#define HAVE_WMMA_F32 1
typedef __attribute__((ext_vector_type(2))) float v2f;
#else
#define HAVE_WMMA_F32 0
#endif

#define ACT_RELU  0
#define ACT_LR02  1
#define ACT_LR01  2
#define ACT_NONE  3

// ---------------------------------------------------------------------------
// Fragment-layout helpers.
// A (16x32 f16): VGPR v holds K = (v&3)*2 + hf*8 + (v>>2)*16 (+1), lane = hf*16+m.
//   inverse of col c: par=c&1, hf=(c>>3)&1, v=(c>>4)*4+((c&7)>>1), idx=2v+par.
// B (32x16 f16): same K order as A, lane = hf*16+m.
//   inverse of row r: hf=(r>>3)&1, idx=((r>>4)<<3)|(r&7).
// Fragments stored per (tile, lane): region[tile*32 + lane][0..15] contiguous,
// so one lane's fragment is a single 32B LDS read (2x ds_load_b128).
// ---------------------------------------------------------------------------
__device__ __forceinline__ int fragA_idx(int c) {
    return (((c >> 4) * 4 + ((c & 7) >> 1)) << 1) | (c & 1);
}
__device__ __forceinline__ int fragA_lane(int c, int m) { return (((c >> 3) & 1) << 4) | m; }
__device__ __forceinline__ int fragB_idx(int r) { return ((r >> 4) << 3) | (r & 7); }
__device__ __forceinline__ int fragB_lane(int r, int m) { return (((r >> 3) & 1) << 4) | m; }

__device__ __forceinline__ void store_tile(float* __restrict__ Z, const v8f& acc,
    const float* __restrict__ bias, int bz, int Ctot, int oOff, int Cout, size_t L,
    int obase, int lbase, int m, int hf)
{
    #pragma unroll
    for (int r = 0; r < 8; ++r) {
        int o = obase + r + hf * 8;      // C/D layout: VGPR r -> M=r / r+8
        if (o < Cout)
            Z[((size_t)bz * Ctot + (oOff + o)) * L + (lbase + m)] = acc[r] + bias[o];
    }
}

// ============================= WMMA GEMM ===================================
// Z[b, oOff+o, l] = sum_c W[o,c] * X[b,c,l] + bias[o]
// 128 threads / 4 waves; block tile 64(o) x 64(l); each wave: 2x2 WMMA tiles.
// Requires L % 64 == 0 (true for every layer in this network) and 16B-aligned X.
__global__ __launch_bounds__(128) void k_gemm_wmma(
    const float* __restrict__ W, const float* __restrict__ bias,
    const float* __restrict__ X, float* __restrict__ Z,
    int Cout, int Cin, int L, int Ctot, int oOff)
{
    __shared__ __align__(32) _Float16 fA[4 * 32 * 16];   // 4 o-tiles, frag layout
    __shared__ __align__(32) _Float16 fB[4 * 32 * 16];   // 4 l-tiles, frag layout
    const int tid  = threadIdx.x;
    const int lane = tid & 31;
    const int wave = tid >> 5;
    const int l0 = blockIdx.x * 64;
    const int o0 = blockIdx.y * 64;
    const int bz = blockIdx.z;
    const size_t xb = (size_t)bz * Cin * L;
    const int m  = lane & 15;
    const int hf = lane >> 4;
    const int wo = (wave >> 1) * 32;     // wave's o offset in block tile
    const int wl = (wave & 1) * 32;      // wave's l offset in block tile
    v8f acc00 = {}, acc01 = {}, acc10 = {}, acc11 = {};
    for (int k0 = 0; k0 < Cin; k0 += 32) {
        __builtin_prefetch(X + xb + (size_t)(k0 + 32) * L + l0, 0, 1);
        // stage W (64 x 32) directly into A-fragment layout
        for (int e = tid; e < 64 * 32; e += 128) {
            int r = e >> 5, c = e & 31;
            float wv = 0.f;
            if (o0 + r < Cout && k0 + c < Cin) wv = W[(size_t)(o0 + r) * Cin + (k0 + c)];
            fA[(((r >> 4) * 32) + fragA_lane(c, r & 15)) * 16 + fragA_idx(c)] = (_Float16)wv;
        }
        // stage X (32 x 64) into B-fragment layout; vector b128 loads along l
        for (int e = tid; e < 32 * 16; e += 128) {
            int r = e >> 4, c4 = (e & 15) << 2;
            v4f xv = {};
            if (k0 + r < Cin) xv = *(const v4f*)&X[xb + (size_t)(k0 + r) * L + (l0 + c4)];
            #pragma unroll
            for (int q = 0; q < 4; ++q) {
                int c = c4 + q;
                fB[(((c >> 4) * 32) + fragB_lane(r, c & 15)) * 16 + fragB_idx(r)] = (_Float16)xv[q];
            }
        }
        __syncthreads();
        const v16h* pA = (const v16h*)fA;
        const v16h* pB = (const v16h*)fB;
        v16h a0 = pA[((wo >> 4) + 0) * 32 + lane];
        v16h a1 = pA[((wo >> 4) + 1) * 32 + lane];
        v16h b0 = pB[((wl >> 4) + 0) * 32 + lane];
        v16h b1 = pB[((wl >> 4) + 1) * 32 + lane];
        acc00 = __builtin_amdgcn_wmma_f32_16x16x32_f16(false, a0, false, b0, (short)0, acc00, false, false);
        acc01 = __builtin_amdgcn_wmma_f32_16x16x32_f16(false, a0, false, b1, (short)0, acc01, false, false);
        acc10 = __builtin_amdgcn_wmma_f32_16x16x32_f16(false, a1, false, b0, (short)0, acc10, false, false);
        acc11 = __builtin_amdgcn_wmma_f32_16x16x32_f16(false, a1, false, b1, (short)0, acc11, false, false);
        __syncthreads();
    }
    store_tile(Z, acc00, bias, bz, Ctot, oOff, Cout, (size_t)L, o0 + wo +  0, l0 + wl +  0, m, hf);
    store_tile(Z, acc01, bias, bz, Ctot, oOff, Cout, (size_t)L, o0 + wo +  0, l0 + wl + 16, m, hf);
    store_tile(Z, acc10, bias, bz, Ctot, oOff, Cout, (size_t)L, o0 + wo + 16, l0 + wl +  0, m, hf);
    store_tile(Z, acc11, bias, bz, Ctot, oOff, Cout, (size_t)L, o0 + wo + 16, l0 + wl + 16, m, hf);
}

// ====================== WMMA GEMM with TDM staging =========================
#if HAVE_TDM
typedef __attribute__((ext_vector_type(4))) unsigned int v4u;
typedef __attribute__((ext_vector_type(8))) int v8i_;
typedef __attribute__((ext_vector_type(4))) int v4i_;

__device__ __forceinline__ void tdm_load_tile_f32(unsigned ldsOff, const float* gptr,
                                                  int rowStride, int dim0, int dim1,
                                                  int tile0, int tile1)
{
    // D# per CDNA5 ISA 8.3/8.4: 2D f32 tile (tile1 rows x tile0 cols)
    unsigned long long ga = (unsigned long long)(uintptr_t)gptr;
    v4u g0;
    g0.x = 1u;                                                // count=1
    g0.y = ldsOff;                                            // lds_addr
    g0.z = (unsigned)(ga & 0xffffffffu);                      // global_addr lo
    g0.w = (unsigned)((ga >> 32) & 0x01ffffffu) | (2u << 30); // global_addr hi | type=2
    v8i_ g1;
    g1[0] = (2 << 16);                                        // wg_mask=0, data_size=4B
    g1[1] = (int)(((unsigned)dim0 & 0xffffu) << 16);          // tensor_dim0 lo
    g1[2] = (int)((((unsigned)dim0 >> 16) & 0xffffu)
            | (((unsigned)dim1 & 0xffffu) << 16));            // dim0 hi | tensor_dim1 lo
    g1[3] = (int)((((unsigned)dim1 >> 16) & 0xffffu)
            | ((unsigned)tile0 << 16));                       // dim1 hi | tile_dim0
    g1[4] = tile1;                                            // tile_dim1, tile_dim2=0
    g1[5] = rowStride;                                        // tensor_dim0_stride lo
    g1[6] = 0;
    g1[7] = 0;
    v4i_ z4 = {0, 0, 0, 0};
#if defined(__clang_major__) && __clang_major__ >= 23
    v8i_ z8 = {0, 0, 0, 0, 0, 0, 0, 0};
    __builtin_amdgcn_tensor_load_to_lds(g0, g1, z4, z4, z8, 0);
#else
    __builtin_amdgcn_tensor_load_to_lds(g0, g1, z4, z4, 0);
#endif
}

__global__ __launch_bounds__(128) void k_gemm_wmma_tdm(
    const float* __restrict__ W, const float* __restrict__ bias,
    const float* __restrict__ X, float* __restrict__ Z,
    int Cout, int Cin, int L, int Ctot, int oOff)
{
    __shared__ __align__(32) _Float16 fA[4 * 32 * 16];
    __shared__ __align__(32) _Float16 fB[4 * 32 * 16];
    __shared__ __align__(16) float    sXf[32 * 64];   // TDM target (f32, k x l)
    const int tid  = threadIdx.x;
    const int lane = tid & 31;
    const int wave = tid >> 5;
    const int l0 = blockIdx.x * 64;
    const int o0 = blockIdx.y * 64;
    const int bz = blockIdx.z;
    const int m  = lane & 15;
    const int hf = lane >> 4;
    const int wo = (wave >> 1) * 32;
    const int wl = (wave & 1) * 32;
    v8f acc00 = {}, acc01 = {}, acc10 = {}, acc11 = {};
    for (int k0 = 0; k0 < Cin; k0 += 32) {
        if (tid == 0) {   // one TDM DMA per block iteration (EXEC ignored by TDM)
            unsigned ldsOff = (unsigned)(unsigned long long)(void*)&sXf[0];
            const float* gp = X + (size_t)bz * Cin * L + (size_t)k0 * L + l0;
            tdm_load_tile_f32(ldsOff, gp, L, L - l0, Cin - k0, 64, 32);
            __builtin_amdgcn_s_wait_tensorcnt(0);
        }
        for (int e = tid; e < 64 * 32; e += 128) {
            int r = e >> 5, c = e & 31;
            float wv = 0.f;
            if (o0 + r < Cout && k0 + c < Cin) wv = W[(size_t)(o0 + r) * Cin + (k0 + c)];
            fA[(((r >> 4) * 32) + fragA_lane(c, r & 15)) * 16 + fragA_idx(c)] = (_Float16)wv;
        }
        __syncthreads();
        for (int e = tid; e < 32 * 16; e += 128) {
            int r = e >> 4, c4 = (e & 15) << 2;
            v4f xv = *(const v4f*)&sXf[(r << 6) | c4];
            #pragma unroll
            for (int q = 0; q < 4; ++q) {
                int c = c4 + q;
                fB[(((c >> 4) * 32) + fragB_lane(r, c & 15)) * 16 + fragB_idx(r)] = (_Float16)xv[q];
            }
        }
        __syncthreads();
        const v16h* pA = (const v16h*)fA;
        const v16h* pB = (const v16h*)fB;
        v16h a0 = pA[((wo >> 4) + 0) * 32 + lane];
        v16h a1 = pA[((wo >> 4) + 1) * 32 + lane];
        v16h b0 = pB[((wl >> 4) + 0) * 32 + lane];
        v16h b1 = pB[((wl >> 4) + 1) * 32 + lane];
        acc00 = __builtin_amdgcn_wmma_f32_16x16x32_f16(false, a0, false, b0, (short)0, acc00, false, false);
        acc01 = __builtin_amdgcn_wmma_f32_16x16x32_f16(false, a0, false, b1, (short)0, acc01, false, false);
        acc10 = __builtin_amdgcn_wmma_f32_16x16x32_f16(false, a1, false, b0, (short)0, acc10, false, false);
        acc11 = __builtin_amdgcn_wmma_f32_16x16x32_f16(false, a1, false, b1, (short)0, acc11, false, false);
        __syncthreads();
    }
    store_tile(Z, acc00, bias, bz, Ctot, oOff, Cout, (size_t)L, o0 + wo +  0, l0 + wl +  0, m, hf);
    store_tile(Z, acc01, bias, bz, Ctot, oOff, Cout, (size_t)L, o0 + wo +  0, l0 + wl + 16, m, hf);
    store_tile(Z, acc10, bias, bz, Ctot, oOff, Cout, (size_t)L, o0 + wo + 16, l0 + wl +  0, m, hf);
    store_tile(Z, acc11, bias, bz, Ctot, oOff, Cout, (size_t)L, o0 + wo + 16, l0 + wl + 16, m, hf);
}
#endif  // HAVE_TDM

// ============================ BatchNorm ====================================
__global__ __launch_bounds__(256) void k_bn_stats(
    const float* __restrict__ Z, float* __restrict__ stats,
    int Cout, int L, int B, int Ctot, int oOff)
{
    __shared__ float s1[256], s2[256];
    int o = blockIdx.x;
    float a = 0.f, b = 0.f;
    for (int i = threadIdx.x; i < B * L; i += 256) {
        int bz = i / L, l = i - bz * L;
        float v = Z[((size_t)bz * Ctot + oOff + o) * L + l];
        a += v; b += v * v;
    }
    s1[threadIdx.x] = a; s2[threadIdx.x] = b;
    __syncthreads();
    for (int st = 128; st > 0; st >>= 1) {
        if (threadIdx.x < st) {
            s1[threadIdx.x] += s1[threadIdx.x + st];
            s2[threadIdx.x] += s2[threadIdx.x + st];
        }
        __syncthreads();
    }
    if (threadIdx.x == 0) {
        float n  = (float)(B * L);
        float mu = s1[0] / n;
        float va = s2[0] / n - mu * mu;
        stats[2 * o]     = mu;
        stats[2 * o + 1] = rsqrtf(fmaxf(va, 0.f) + 1e-5f);
    }
}

__global__ void k_bn_act(float* __restrict__ Z, const float* __restrict__ stats,
                         const float* __restrict__ g, const float* __restrict__ bt,
                         int Cout, int L, int B, int Ctot, int oOff, int act)
{
    size_t i = (size_t)blockIdx.x * blockDim.x + threadIdx.x;
    size_t tot = (size_t)B * Cout * L;
    if (i >= tot) return;
    int l = (int)(i % L); size_t t = i / L;
    int o = (int)(t % Cout); int bz = (int)(t / Cout);
    size_t p = ((size_t)bz * Ctot + oOff + o) * L + l;
    float v = Z[p];
    v = g[o] * (v - stats[2 * o]) * stats[2 * o + 1] + bt[o];
    if      (act == ACT_RELU) v = fmaxf(v, 0.f);
    else if (act == ACT_LR02) v = v > 0.f ? v : 0.2f * v;
    else if (act == ACT_LR01) v = v > 0.f ? v : 0.01f * v;
    Z[p] = v;
}

// =============================== KNN =======================================
// 4 waves/block, each wave: 16 query rows; Gram tiles via WMMA; top-16 in regs.
// f32 path: one V_WMMA_F32_16X16X4_F32 per 16x16 tile (exact, matches ref).
__global__ __launch_bounds__(128) void k_knn(
    const float* __restrict__ coords, const _Float16* __restrict__ ch,
    const float* __restrict__ sq,
    int* __restrict__ idxOut, float* __restrict__ d2Out, int N)
{
    __shared__ float sT[4][16][17];
    const int lane = threadIdx.x & 31;
    const int wave = threadIdx.x >> 5;
    const int bz = blockIdx.y;
    const int q0 = blockIdx.x * 64 + wave * 16;
    const int m  = lane & 15;
    const int hf = lane >> 4;
    const float* sb = sq + (size_t)bz * N;

#if HAVE_WMMA_F32
    const float* cf = coords + (size_t)bz * N * 3;
    // A 16x4 f32 layout: VGPR0 = K0 (hf=0) / K2 (hf=1); VGPR1 = K1 / K3
    v2f a;
    {
        float c0 = cf[(size_t)(q0 + m) * 3 + 0];
        float c1 = cf[(size_t)(q0 + m) * 3 + 1];
        float c2 = cf[(size_t)(q0 + m) * 3 + 2];
        a.x = hf ? c2 : c0;
        a.y = hf ? 0.f : c1;
    }
#else
    const _Float16* cb = ch + (size_t)bz * N * 4;
    v16h a = {};
    if (hf == 0) {
        a[0] = cb[(size_t)(q0 + m) * 4 + 0];
        a[1] = cb[(size_t)(q0 + m) * 4 + 1];
        a[2] = cb[(size_t)(q0 + m) * 4 + 2];
    }
#endif
    float bd[16]; int bi[16];
    #pragma unroll
    for (int t = 0; t < 16; ++t) { bd[t] = 3.4e38f; bi[t] = 0; }

    for (int j0 = 0; j0 < N; j0 += 16) {
        v8f acc = {};
#if HAVE_WMMA_F32
        v2f b;
        {
            float c0 = cf[(size_t)(j0 + m) * 3 + 0];
            float c1 = cf[(size_t)(j0 + m) * 3 + 1];
            float c2 = cf[(size_t)(j0 + m) * 3 + 2];
            b.x = hf ? c2 : c0;
            b.y = hf ? 0.f : c1;
        }
        acc = __builtin_amdgcn_wmma_f32_16x16x4_f32(false, a, false, b, (short)0, acc, false, false);
#else
        v16h b = {};
        if (hf == 0) {
            b[0] = cb[(size_t)(j0 + m) * 4 + 0];
            b[1] = cb[(size_t)(j0 + m) * 4 + 1];
            b[2] = cb[(size_t)(j0 + m) * 4 + 2];
        }
        acc = __builtin_amdgcn_wmma_f32_16x16x32_f16(false, a, false, b, (short)0, acc, false, false);
#endif
        #pragma unroll
        for (int r = 0; r < 8; ++r) {
            int qi = r + hf * 8;
            int j  = j0 + m;
            float d2 = sb[q0 + qi] + sb[j] - 2.f * acc[r];
            if (j == q0 + qi) d2 = 3.4e38f;   // exclude self (ref drops top-1)
            sT[wave][qi][m] = d2;
        }
        __syncthreads();
        if (lane < 16) {
            for (int n = 0; n < 16; ++n) {
                float d = sT[wave][lane][n];
                if (d < bd[15]) {
                    int p = 15;
                    #pragma unroll
                    for (int t = 15; t > 0; --t)
                        if (d < bd[t - 1]) { bd[t] = bd[t - 1]; bi[t] = bi[t - 1]; p = t - 1; }
                    bd[p] = d; bi[p] = j0 + n;
                }
            }
        }
        __syncthreads();
    }
    if (lane < 16) {
        int q = q0 + lane;
        for (int t = 0; t < 16; ++t) {
            idxOut[((size_t)bz * N + q) * 16 + t] = bi[t];
            d2Out [((size_t)bz * N + q) * 16 + t] = fmaxf(bd[t], 0.f);  // ref keeps d^2
        }
    }
}

// ====================== elementwise / gather kernels =======================
__global__ void k_coords_from_x(const float* __restrict__ x, float* __restrict__ coords, int N, int B)
{
    int i = blockIdx.x * blockDim.x + threadIdx.x;   // over B*N
    if (i >= B * N) return;
    coords[(size_t)i * 3 + 0] = x[(size_t)i * 6 + 0];
    coords[(size_t)i * 3 + 1] = x[(size_t)i * 6 + 1];
    coords[(size_t)i * 3 + 2] = x[(size_t)i * 6 + 2];
}

__global__ void k_coords_aux(const float* __restrict__ coords, _Float16* __restrict__ ch,
                             float* __restrict__ sq, int N, int B)
{
    int i = blockIdx.x * blockDim.x + threadIdx.x;   // over B*N
    if (i >= B * N) return;
    float a = coords[(size_t)i * 3 + 0];
    float b = coords[(size_t)i * 3 + 1];
    float c = coords[(size_t)i * 3 + 2];
    _Float16* o = ch + (size_t)i * 4;
    o[0] = (_Float16)a; o[1] = (_Float16)b; o[2] = (_Float16)c; o[3] = (_Float16)0.f;
    sq[i] = a * a + b * b + c * c;
}

__global__ void k_fcstart(const float* __restrict__ x, const float* __restrict__ W,
                          const float* __restrict__ b, float* __restrict__ Z, int N, int B)
{
    size_t i = (size_t)blockIdx.x * blockDim.x + threadIdx.x;   // over B*8*N
    if (i >= (size_t)B * 8 * N) return;
    int n = (int)(i % N); size_t t = i / N;
    int o = (int)(t % 8); int bz = (int)(t / 8);
    float s = b[o];
    for (int d = 0; d < 6; ++d) s += x[((size_t)bz * N + n) * 6 + d] * W[o * 6 + d];
    Z[((size_t)bz * 8 + o) * N + n] = s;
}

__global__ void k_lse_sp(const float* __restrict__ coords, const int* __restrict__ idx,
                         const float* __restrict__ d2, float* __restrict__ sp, int N, int B)
{
    size_t i = (size_t)blockIdx.x * blockDim.x + threadIdx.x;   // over B*N*16
    if (i >= (size_t)B * N * 16) return;
    int k = (int)(i & 15); size_t t = i >> 4;
    int n = (int)(t % N); int bz = (int)(t / N);
    int j = idx[((size_t)bz * N + n) * 16 + k];
    const float* cc = coords + ((size_t)bz * N + n) * 3;
    const float* cj = coords + ((size_t)bz * N + j) * 3;
    size_t Lk = (size_t)N * 16, l = (size_t)n * 16 + k;
    float* o = sp + (size_t)bz * 10 * Lk;
    float c0 = cc[0], c1 = cc[1], c2 = cc[2];
    float j0 = cj[0], j1 = cj[1], j2 = cj[2];
    o[0 * Lk + l] = c0;      o[1 * Lk + l] = c1;      o[2 * Lk + l] = c2;
    o[3 * Lk + l] = j0;      o[4 * Lk + l] = j1;      o[5 * Lk + l] = j2;
    o[6 * Lk + l] = c0 - j0; o[7 * Lk + l] = c1 - j1; o[8 * Lk + l] = c2 - j2;
    o[9 * Lk + l] = d2[((size_t)bz * N + n) * 16 + k];
}

__global__ void k_bcast_feat(const float* __restrict__ f, float* __restrict__ out,
                             int Cf, int N, int B, int Ctot, int off)
{
    size_t i = (size_t)blockIdx.x * blockDim.x + threadIdx.x;   // over B*Cf*N*16
    if (i >= (size_t)B * Cf * N * 16) return;
    int k = (int)(i & 15); size_t t = i >> 4;
    int n = (int)(t % N); t /= N;
    int c = (int)(t % Cf); int bz = (int)(t / Cf);
    out[(((size_t)bz * Ctot + off + c) * N + n) * 16 + k] = f[((size_t)bz * Cf + c) * N + n];
}

// softmax over K=16 of precomputed logits, then weighted sum of x over K.
__global__ void k_attpool_apply(const float* __restrict__ x, const float* __restrict__ lgt,
                                float* __restrict__ out, int C, int N, int B)
{
    size_t i = (size_t)blockIdx.x * blockDim.x + threadIdx.x;   // over B*C*N
    if (i >= (size_t)B * C * N) return;
    int n = (int)(i % N); size_t t = i / N;
    int d = (int)(t % C); int bz = (int)(t / C);
    size_t Lk = (size_t)N * 16;
    const float* xb = x   + ((size_t)bz * C + d) * Lk + (size_t)n * 16;
    const float* lb = lgt + ((size_t)bz * C + d) * Lk + (size_t)n * 16;
    float lg[16];
    #pragma unroll
    for (int k = 0; k < 16; ++k) lg[k] = lb[k];
    float mx = lg[0];
    #pragma unroll
    for (int k = 1; k < 16; ++k) mx = fmaxf(mx, lg[k]);
    float sm = 0.f;
    #pragma unroll
    for (int k = 0; k < 16; ++k) { lg[k] = __expf(lg[k] - mx); sm += lg[k]; }
    float o = 0.f;
    #pragma unroll
    for (int k = 0; k < 16; ++k) o += lg[k] * xb[k];
    out[((size_t)bz * C + d) * N + n] = o / sm;
}

__global__ void k_leaky(float* __restrict__ p, size_t tot, float slope)
{
    size_t i = (size_t)blockIdx.x * blockDim.x + threadIdx.x;
    if (i >= tot) return;
    float v = p[i];
    p[i] = v > 0.f ? v : slope * v;
}

// out[b][oOff+c][i] = in[b][c][ i*inN/outN ]  (decimation, interp, concat-copy)
__global__ void k_gather_cols(const float* __restrict__ in, float* __restrict__ out,
                              int C, int inN, int outN, int B, int Ctot, int oOff)
{
    size_t i = (size_t)blockIdx.x * blockDim.x + threadIdx.x;   // over B*C*outN
    if (i >= (size_t)B * C * outN) return;
    int n = (int)(i % outN); size_t t = i / outN;
    int c = (int)(t % C); int bz = (int)(t / C);
    int src = (int)((long long)n * inN / outN);
    out[((size_t)bz * Ctot + oOff + c) * outN + n] = in[((size_t)bz * C + c) * inN + src];
}

__global__ void k_gather_coords(const float* __restrict__ in, float* __restrict__ out,
                                int inN, int outN, int B)
{
    int i = blockIdx.x * blockDim.x + threadIdx.x;   // over B*outN*3
    if (i >= B * outN * 3) return;
    int c = i % 3; int t = i / 3;
    int n = t % outN; int bz = t / outN;
    int src = (int)((long long)n * inN / outN);
    out[((size_t)bz * outN + n) * 3 + c] = in[((size_t)bz * inN + src) * 3 + c];
}

// ============================== host side ==================================
struct Smlp { const float *W, *b, *g, *bt; };

static inline unsigned cdiv(size_t a, unsigned b) { return (unsigned)((a + b - 1) / b); }

static void run_smlp(hipStream_t s, Smlp p, const float* X, float* Z, float* stats,
                     int Cout, int Cin, int L, int B, int Ctot, int oOff, int act, bool tdm)
{
    dim3 gg(cdiv((size_t)L, 64), cdiv((size_t)Cout, 64), (unsigned)B);
    bool launched = false;
#if HAVE_TDM
    if (tdm) {
        k_gemm_wmma_tdm<<<gg, 128, 0, s>>>(p.W, p.b, X, Z, Cout, Cin, L, Ctot, oOff);
        launched = true;
    }
#endif
    if (!launched)
        k_gemm_wmma<<<gg, 128, 0, s>>>(p.W, p.b, X, Z, Cout, Cin, L, Ctot, oOff);
    k_bn_stats<<<Cout, 256, 0, s>>>(Z, stats, Cout, L, B, Ctot, oOff);
    size_t tot = (size_t)B * Cout * L;
    k_bn_act<<<cdiv(tot, 256), 256, 0, s>>>(Z, stats, p.g, p.bt, Cout, L, B, Ctot, oOff, act);
    (void)tdm;
}

// attention pool: logits via WMMA GEMM (square W, bias), softmax+sum applied after
static void run_attpool(hipStream_t s, const float* W, const float* bias,
                        const float* x, float* logits, float* out, int C, int n, int B)
{
    size_t Lk = (size_t)n * 16;
    dim3 gg(cdiv(Lk, 64), cdiv((size_t)C, 64), (unsigned)B);
    k_gemm_wmma<<<gg, 128, 0, s>>>(W, bias, x, logits, C, C, (int)Lk, C, 0);
    k_attpool_apply<<<cdiv((size_t)B * C * n, 256), 256, 0, s>>>(x, logits, out, C, n, B);
}

extern "C" void kernel_launch(void* const* d_in, const int* in_sizes, int n_in,
                              void* d_out, int out_size, void* d_ws, size_t ws_size,
                              hipStream_t stream)
{
    (void)in_sizes; (void)n_in; (void)out_size; (void)ws_size;
    const int B = 2, N0 = 8192;
    auto F = [&](int i) { return (const float*)d_in[i]; };
    auto smlpAt = [&](int base) { Smlp s{F(base), F(base + 1), F(base + 2), F(base + 3)}; return s; };

    // --- assumed param flattening (dict insertion order) ---
    const float* fcW = F(1); const float* fcB = F(2);
    const float* bng = F(3); const float* bnb = F(4);
    // enc lfa blocks: 28 arrays each, starting at index 5
    // dec: 117..132 ; fc_end: 133..144

    // --- workspace carving ---
    char* wp = (char*)d_ws;
    auto alloc = [&](size_t elems) -> float* {
        float* r = (float*)wp;
        wp += ((elems * sizeof(float) + 255) & ~(size_t)255);
        return r;
    };
    float* hA      = alloc(2 * 1048576);
    float* hB      = alloc(2 * 1048576);
    float* coordsA = alloc((size_t)B * N0 * 3);
    float* coordsB = alloc((size_t)B * N0 * 3);
    _Float16* chH  = (_Float16*)alloc((size_t)B * N0 * 2);   // B*N0*4 halves
    float* sq      = alloc((size_t)B * N0);
    int*   idxB    = (int*)alloc((size_t)B * N0 * 16);
    float* distB   = alloc((size_t)B * N0 * 16);
    float* sp      = alloc((size_t)B * 10 * N0 * 16);
    float* cat1    = alloc((size_t)B * 16 * N0 * 16);        // max d_out*N*K (level 0)
    float* cat2    = alloc((size_t)B * 8 * N0 * 16);
    float* logitsB = alloc((size_t)B * 16 * N0 * 16);        // attpool logits scratch
    float* pooled  = alloc((size_t)B * 16 * N0);
    float* pooled2 = alloc((size_t)B * 16 * N0);
    float* bufm1   = alloc((size_t)B * 16 * N0);
    float* bufp    = alloc((size_t)B * 16 * N0);
    float* stats   = alloc(2048);
    const int dout_[4]  = {16, 64, 128, 256};
    const int chIn_[4]  = {8, 32, 128, 256};
    const int nLvl[5]   = {8192, 2048, 512, 128, 32};
    float* xs[4];
    for (int i = 0; i < 4; ++i) xs[i] = alloc((size_t)B * 2 * dout_[i] * nLvl[i]);

    // --- fc_start + bn_start + lrelu0.2 ---
    k_coords_from_x<<<cdiv((size_t)B * N0, 256), 256, 0, stream>>>(F(0), coordsA, N0, B);
    k_fcstart<<<cdiv((size_t)B * 8 * N0, 256), 256, 0, stream>>>(F(0), fcW, fcB, hA, N0, B);
    k_bn_stats<<<8, 256, 0, stream>>>(hA, stats, 8, N0, B, 8, 0);
    k_bn_act<<<cdiv((size_t)B * 8 * N0, 256), 256, 0, stream>>>(hA, stats, bng, bnb, 8, N0, B, 8, 0, ACT_LR02);

    float* h = hA; float* hO = hB;
    float* co = coordsA; float* coO = coordsB;
    int hCh = 8;

    // ------------------------------ encoder -------------------------------
    for (int li = 0; li < 4; ++li) {
        const int n  = nLvl[li];
        const int dO = dout_[li];
        const int Fh = dO / 2;
        const int Q  = dO / 4;
        const int ch2 = 2 * dO;
        const int Cin = chIn_[li];
        const size_t Lk = (size_t)n * 16;
        const int base = 5 + li * 28;
        Smlp mlp1 = smlpAt(base + 0), lse1 = smlpAt(base + 4);
        const float* p1W = F(base + 8); const float* p1b = F(base + 9);
        Smlp mlpp = smlpAt(base + 10), lse2 = smlpAt(base + 14);
        const float* p2W = F(base + 18); const float* p2b = F(base + 19);
        Smlp mlp2 = smlpAt(base + 20), resd = smlpAt(base + 24);

        // knn at this level (WMMA Gram tiles, f32 path when available)
        k_coords_aux<<<cdiv((size_t)B * n, 256), 256, 0, stream>>>(co, chH, sq, n, B);
        k_knn<<<dim3((unsigned)(n / 64), (unsigned)B), 128, 0, stream>>>(co, chH, sq, idxB, distB, n);

        // mlp1 (lrelu 0.2)
        run_smlp(stream, mlp1, h, bufm1, stats, Fh, Cin, n, B, Fh, 0, ACT_LR02, false);
        // sp features (shared by lse1/lse2)
        k_lse_sp<<<cdiv((size_t)B * n * 16, 256), 256, 0, stream>>>(co, idxB, distB, sp, n, B);
        // lse1: smlp(sp) -> cat1[0:Fh), broadcast mlp1 -> cat1[Fh:2Fh)
        run_smlp(stream, lse1, sp, cat1, stats, Fh, 10, (int)Lk, B, dO, 0, ACT_RELU, false);
        k_bcast_feat<<<cdiv((size_t)B * Fh * Lk, 256), 256, 0, stream>>>(bufm1, cat1, Fh, n, B, dO, Fh);
        // pool1 (WMMA logits + softmax apply) + mlp_pool1
        run_attpool(stream, p1W, p1b, cat1, logitsB, pooled, dO, n, B);
        run_smlp(stream, mlpp, pooled, bufp, stats, Q, dO, n, B, Q, 0, ACT_RELU, false);
        // lse2 -> cat2
        run_smlp(stream, lse2, sp, cat2, stats, Q, 10, (int)Lk, B, 2 * Q, 0, ACT_RELU, false);
        k_bcast_feat<<<cdiv((size_t)B * Q * Lk, 256), 256, 0, stream>>>(bufp, cat2, Q, n, B, 2 * Q, Q);
        // pool2 + mlp2 -> xs[li][0:dO)
        run_attpool(stream, p2W, p2b, cat2, logitsB, pooled2, 2 * Q, n, B);
        run_smlp(stream, mlp2, pooled2, xs[li], stats, dO, 2 * Q, n, B, ch2, 0, ACT_RELU, false);
        // residual -> xs[li][dO:2dO), then lrelu 0.01
        run_smlp(stream, resd, h, xs[li], stats, dO, Cin, n, B, ch2, dO, ACT_RELU, false);
        k_leaky<<<cdiv((size_t)B * ch2 * n, 256), 256, 0, stream>>>(xs[li], (size_t)B * ch2 * n, 0.01f);

        // decimate (deterministic stride; ref uses a host-side random permutation)
        const int nd = nLvl[li + 1];
        k_gather_cols<<<cdiv((size_t)B * ch2 * nd, 256), 256, 0, stream>>>(xs[li], hO, ch2, n, nd, B, ch2, 0);
        k_gather_coords<<<cdiv((size_t)B * nd * 3, 256), 256, 0, stream>>>(co, coO, n, nd, B);
        float* t0 = h; h = hO; hO = t0;
        float* t1 = co; co = coO; coO = t1;
        hCh = ch2;
    }

    // ------------------------------ decoder -------------------------------
    const int decOut[4] = {256, 128, 32, 8};
    int curN = nLvl[4];
    for (int di = 0; di < 4; ++di) {
        const int upN = nLvl[3 - di];
        const int Cs  = 2 * dout_[3 - di];
        const int Ct  = hCh + Cs;
        const int Co  = decOut[di];
        Smlp dp = smlpAt(117 + 4 * di);
        k_gather_cols<<<cdiv((size_t)B * hCh * upN, 256), 256, 0, stream>>>(h, cat1, hCh, curN, upN, B, Ct, 0);
        k_gather_cols<<<cdiv((size_t)B * Cs * upN, 256), 256, 0, stream>>>(xs[3 - di], cat1, Cs, nLvl[3 - di], upN, B, Ct, hCh);
        run_smlp(stream, dp, cat1, hO, stats, Co, Ct, upN, B, Co, 0, ACT_RELU, true);
        float* t0 = h; h = hO; hO = t0;
        hCh = Co; curN = upN;
    }

    // ------------------------------ fc_end --------------------------------
    Smlp f0 = smlpAt(133), f1 = smlpAt(137), f2 = smlpAt(141);
    run_smlp(stream, f0, h, hO, stats, 64, 8, curN, B, 64, 0, ACT_RELU, true);
    { float* t0 = h; h = hO; hO = t0; }
    run_smlp(stream, f1, h, hO, stats, 32, 64, curN, B, 32, 0, ACT_RELU, true);
    { float* t0 = h; h = hO; hO = t0; }
    run_smlp(stream, f2, h, (float*)d_out, stats, 13, 32, curN, B, 13, 0, ACT_RELU, true);
}
